// GNN_Graph_2362232012849
// MI455X (gfx1250) — matrix operand order repaired and checked by reading, not compiled, not run on hardware
//
#include <hip/hip_runtime.h>
#include <stdint.h>

#define N_NODES 50000
#define N_EDGES 800000
#define N_GRAPH 256
#define DIM     300
#define NLAYER  4
#define EDIM    50
#define GFDIM   10
#define HDIM    (DIM + GFDIM)   /* 310 */
#define BN_EPS  1e-5f

typedef __attribute__((ext_vector_type(16))) __bf16 v16bf;
typedef __attribute__((ext_vector_type(8)))  __bf16 v8bf;
typedef __attribute__((ext_vector_type(8)))  float  v8f;

__device__ __forceinline__ uint16_t f2bfbits(float f) {
  uint32_t b = __float_as_uint(f);
  b += 0x7FFFu + ((b >> 16) & 1u);          // round-to-nearest-even
  return (uint16_t)(b >> 16);
}
__device__ __forceinline__ __bf16 f2bf16(float f) {
  uint16_t s = f2bfbits(f);
  return __builtin_bit_cast(__bf16, s);
}
__device__ __forceinline__ unsigned enc_f(float f) {
  unsigned b = __float_as_uint(f);
  return (b & 0x80000000u) ? ~b : (b | 0x80000000u);
}
__device__ __forceinline__ float dec_f(unsigned u) {
  unsigned b = (u & 0x80000000u) ? (u & 0x7FFFFFFFu) : ~u;
  return __uint_as_float(b);
}

// ---------------- f32 -> padded bf16 convert ----------------
__global__ void k_pad_bf16(const float* __restrict__ src, uint16_t* __restrict__ dst,
                           int srows, int scols, long long total, int dcols) {
  long long idx = (long long)blockIdx.x * blockDim.x + threadIdx.x;
  if (idx >= total) return;
  int c = (int)(idx % dcols);
  long long r = idx / dcols;
  float v = (r < srows && c < scols) ? src[r * (long long)scols + c] : 0.0f;
  dst[idx] = f2bfbits(v);
}

// ---------------- generic bf16 WMMA GEMM: C = A[M,Kp] * B[Kp,320] (+bias)(+relu) ----
// grid = (M/16), block = 128 (4 waves); each wave owns 5 N-tiles -> 320 padded cols.
// A-fragment loaded once per K-step and reused across all 5 WMMAs.
__global__ void __launch_bounds__(128)
k_gemm_bf16(const uint16_t* __restrict__ A, int lda,
            const uint16_t* __restrict__ B,          // ldb fixed at 320
            const float* __restrict__ bias,
            float* __restrict__ C, int ldo,
            int Kp, int Dout, int relu) {
  const int lane = threadIdx.x & 31;
  const int wid  = threadIdx.x >> 5;
  const int m0   = blockIdx.x << 4;
  const int half = lane >> 4;
  const int mrow = m0 + (lane & 15);

  v8f acc[5];
  #pragma unroll
  for (int j = 0; j < 5; ++j) acc[j] = (v8f){0.f,0.f,0.f,0.f,0.f,0.f,0.f,0.f};

  for (int k0 = 0; k0 < Kp; k0 += 32) {
    const uint16_t* ap = A + (long long)mrow * lda + k0 + half * 8;
    v8bf alo = *(const v8bf*)ap;               // K = k0 + 8*half + 0..7
    v8bf ahi = *(const v8bf*)(ap + 16);        // K = k0 + 16 + 8*half + 0..7
    v16bf a = __builtin_shufflevector(alo, ahi,
              0,1,2,3,4,5,6,7,8,9,10,11,12,13,14,15);
    const uint16_t* bp = B + (long long)k0 * 320 + (lane & 15);
    #pragma unroll
    for (int j = 0; j < 5; ++j) {
      const int n0 = (wid + 4 * j) << 4;
      v16bf b;
      #pragma unroll
      for (int s = 0; s < 8; ++s) {
        b[s]     = __builtin_bit_cast(__bf16, bp[(long long)(half * 8 + s) * 320 + n0]);
        b[s + 8] = __builtin_bit_cast(__bf16, bp[(long long)(16 + half * 8 + s) * 320 + n0]);
      }
      acc[j] = __builtin_amdgcn_wmma_f32_16x16x32_bf16(false, a, false, b,
                                                       (short)0, acc[j], false, false);
    }
  }

  const int mbase = m0 + half * 8;
  #pragma unroll
  for (int j = 0; j < 5; ++j) {
    const int n = ((wid + 4 * j) << 4) + (lane & 15);
    if (n < Dout) {
      const float bv = bias ? bias[n] : 0.0f;
      #pragma unroll
      for (int i = 0; i < 8; ++i) {
        float v = acc[j][i] + bv;
        if (relu) v = fmaxf(v, 0.0f);
        C[(long long)(mbase + i) * ldo + n] = v;
      }
    }
  }
}

// ---------------- fused GINE edge message: agg[dst] += relu(h[src] + EA*W + b) ----
// 16 edges per block, 4 waves, 5 N-tiles each (covers 320 padded cols)
__global__ void __launch_bounds__(128)
k_edge_msg(const float* __restrict__ edge_attr,   // [E, 50] f32
           const uint16_t* __restrict__ WB,       // [64, 320] bf16
           const float* __restrict__ ebias,       // [300]
           const float* __restrict__ h,           // [N, 300]
           const int* __restrict__ src, const int* __restrict__ dst,
           float* __restrict__ agg) {
  const int lane = threadIdx.x & 31;
  const int wid  = threadIdx.x >> 5;
  const int e0   = blockIdx.x << 4;
  const int half = lane >> 4;
  const int mrow = e0 + (lane & 15);

  // Build A fragments (16x64 bf16, padded from 16x50) in-register from f32.
  const float* arow = edge_attr + (long long)mrow * EDIM;
  v16bf a0, a1;
  #pragma unroll
  for (int i = 0; i < 8; ++i) {
    a0[i]     = f2bf16(arow[half * 8 + i]);        // K 0..15
    a0[i + 8] = f2bf16(arow[16 + half * 8 + i]);   // K 16..31
    a1[i]     = f2bf16(arow[32 + half * 8 + i]);   // K 32..47
    int k = 48 + half * 8 + i;                     // K 48..63 (pad >= 50)
    a1[i + 8] = (k < EDIM) ? f2bf16(arow[k]) : f2bf16(0.0f);
  }

  int sv[8], dv[8];
  #pragma unroll
  for (int i = 0; i < 8; ++i) {
    int e = e0 + half * 8 + i;
    sv[i] = src[e];
    dv[i] = dst[e];
  }

  #pragma unroll
  for (int j = 0; j < 5; ++j) {
    const int n0 = (wid + 4 * j) << 4;
    v16bf b0, b1;
    #pragma unroll
    for (int s = 0; s < 8; ++s) {
      const uint16_t* wp = WB + n0 + (lane & 15);
      b0[s]     = __builtin_bit_cast(__bf16, wp[(long long)(half * 8 + s) * 320]);
      b0[s + 8] = __builtin_bit_cast(__bf16, wp[(long long)(16 + half * 8 + s) * 320]);
      b1[s]     = __builtin_bit_cast(__bf16, wp[(long long)(32 + half * 8 + s) * 320]);
      b1[s + 8] = __builtin_bit_cast(__bf16, wp[(long long)(48 + half * 8 + s) * 320]);
    }
    v8f acc = {0.f, 0.f, 0.f, 0.f, 0.f, 0.f, 0.f, 0.f};
    acc = __builtin_amdgcn_wmma_f32_16x16x32_bf16(false, a0, false, b0,
                                                  (short)0, acc, false, false);
    acc = __builtin_amdgcn_wmma_f32_16x16x32_bf16(false, a1, false, b1,
                                                  (short)0, acc, false, false);
    const int n = n0 + (lane & 15);
    if (n < DIM) {
      const float bv = ebias[n];
      #pragma unroll
      for (int i = 0; i < 8; ++i) {
        float v = acc[i] + bv + h[(long long)sv[i] * DIM + n];
        v = fmaxf(v, 0.0f);
        __hip_atomic_fetch_add(&agg[(long long)dv[i] * DIM + n], v,
                               __ATOMIC_RELAXED, __HIP_MEMORY_SCOPE_AGENT);
      }
    }
  }
}

// ---------------- elementwise helpers ----------------
__global__ void k_init_agg(const float* __restrict__ h, float* __restrict__ agg,
                           const float* __restrict__ eps, int l, long long total) {
  long long idx = (long long)blockIdx.x * blockDim.x + threadIdx.x;
  if (idx >= total) return;
  agg[idx] = (1.0f + eps[l]) * h[idx];
}

__global__ void k_bn_relu(const float* __restrict__ z, float* __restrict__ out,
                          const float* __restrict__ gamma, const float* __restrict__ beta,
                          const float* __restrict__ mean, const float* __restrict__ var,
                          long long total, int dorelu) {
  long long idx = (long long)blockIdx.x * blockDim.x + threadIdx.x;
  if (idx >= total) return;
  int c = (int)(idx % DIM);
  float v = (z[idx] - mean[c]) * rsqrtf(var[c] + BN_EPS) * gamma[c] + beta[c];
  if (dorelu) v = fmaxf(v, 0.0f);
  out[idx] = v;
}

// gate tail: gt[i] = relu(BN(t[i,:])) . W2 + b2
__global__ void k_gate2(const float* __restrict__ t,
                        const float* __restrict__ gamma, const float* __restrict__ beta,
                        const float* __restrict__ mean, const float* __restrict__ var,
                        const float* __restrict__ W2, const float* __restrict__ b2,
                        float* __restrict__ gt) {
  int i = blockIdx.x * blockDim.x + threadIdx.x;
  if (i >= N_NODES) return;
  const float* row = t + (long long)i * DIM;
  float s = 0.0f;
  for (int d = 0; d < DIM; ++d) {
    float v = (row[d] - mean[d]) * rsqrtf(var[d] + BN_EPS) * gamma[d] + beta[d];
    v = fmaxf(v, 0.0f);
    s += v * W2[d];
  }
  gt[i] = s + b2[0];
}

__global__ void k_pool_init(float* __restrict__ hg, unsigned* __restrict__ gmaxu,
                            float* __restrict__ den) {
  int idx = blockIdx.x * blockDim.x + threadIdx.x;
  if (idx < N_GRAPH * DIM) hg[idx] = 0.0f;
  if (idx < N_GRAPH) { gmaxu[idx] = 0x007FFFFFu; den[idx] = 0.0f; }  // enc(-inf)
}

__global__ void k_gmax(const float* __restrict__ gt, const int* __restrict__ batch,
                       unsigned* __restrict__ gmaxu) {
  int i = blockIdx.x * blockDim.x + threadIdx.x;
  if (i >= N_NODES) return;
  atomicMax(&gmaxu[batch[i]], enc_f(gt[i]));
}

__global__ void k_ex(const float* __restrict__ gt, const int* __restrict__ batch,
                     const unsigned* __restrict__ gmaxu,
                     float* __restrict__ exv, float* __restrict__ den) {
  int i = blockIdx.x * blockDim.x + threadIdx.x;
  if (i >= N_NODES) return;
  int b = batch[i];
  float e = expf(gt[i] - dec_f(gmaxu[b]));
  exv[i] = e;
  __hip_atomic_fetch_add(&den[b], e, __ATOMIC_RELAXED, __HIP_MEMORY_SCOPE_AGENT);
}

__global__ void k_pool(const float* __restrict__ h, const float* __restrict__ exv,
                       const float* __restrict__ den, const int* __restrict__ batch,
                       float* __restrict__ hg, long long total) {
  long long idx = (long long)blockIdx.x * blockDim.x + threadIdx.x;
  if (idx >= total) return;
  int i = (int)(idx / DIM);
  int d = (int)(idx % DIM);
  int b = batch[i];
  float dn = den[b];
  float sc = exv[i] / (dn > 0.0f ? dn : 1.0f);
  __hip_atomic_fetch_add(&hg[(long long)b * DIM + d], sc * h[idx],
                         __ATOMIC_RELAXED, __HIP_MEMORY_SCOPE_AGENT);
}

__global__ void k_hc(const float* __restrict__ hg, const float* __restrict__ gvec,
                     float* __restrict__ hc) {
  int idx = blockIdx.x * blockDim.x + threadIdx.x;
  if (idx >= N_GRAPH * HDIM) return;
  int g = idx / HDIM, c = idx % HDIM;
  hc[idx] = (c < DIM) ? hg[g * DIM + c] : gvec[g * GFDIM + (c - DIM)];
}

__global__ void k_dense(const float* __restrict__ in, const float* __restrict__ W,
                        const float* __restrict__ b, float* __restrict__ out,
                        int IN, int OUT, int relu) {
  int idx = blockIdx.x * blockDim.x + threadIdx.x;
  if (idx >= N_GRAPH * OUT) return;
  int g = idx / OUT, o = idx % OUT;
  float s = b[o];
  for (int k = 0; k < IN; ++k) s += in[g * IN + k] * W[k * OUT + o];
  if (relu) s = fmaxf(s, 0.0f);
  out[idx] = s;
}

static inline size_t align256(size_t x) { return (x + 255) & ~(size_t)255; }
static inline unsigned gdiv(long long n, int bs) { return (unsigned)((n + bs - 1) / bs); }

extern "C" void kernel_launch(void* const* d_in, const int* in_sizes, int n_in,
                              void* d_out, int out_size, void* d_ws, size_t ws_size,
                              hipStream_t stream) {
  const float* x          = (const float*)d_in[0];
  const int*   edge_index = (const int*)  d_in[1];
  const float* edge_attr  = (const float*)d_in[2];
  const int*   batch      = (const int*)  d_in[3];
  const float* gvec       = (const float*)d_in[4];
  const float* atom_W     = (const float*)d_in[5];
  const float* atom_b     = (const float*)d_in[6];
  const float* edge_W     = (const float*)d_in[7];
  const float* edge_b     = (const float*)d_in[8];
  const float* mlp_W1     = (const float*)d_in[9];
  const float* mlp_b1     = (const float*)d_in[10];
  const float* mlp_W2     = (const float*)d_in[11];
  const float* mlp_b2     = (const float*)d_in[12];
  const float* eps        = (const float*)d_in[13];
  const float* bn_gamma   = (const float*)d_in[14];
  const float* bn_beta    = (const float*)d_in[15];
  const float* bn_mean    = (const float*)d_in[16];
  const float* bn_var     = (const float*)d_in[17];
  const float* gate_W1    = (const float*)d_in[18];
  const float* gate_b1    = (const float*)d_in[19];
  const float* g_bn_gamma = (const float*)d_in[20];
  const float* g_bn_beta  = (const float*)d_in[21];
  const float* g_bn_mean  = (const float*)d_in[22];
  const float* g_bn_var   = (const float*)d_in[23];
  const float* gate_W2    = (const float*)d_in[24];
  const float* gate_b2    = (const float*)d_in[25];
  const float* h0_W1      = (const float*)d_in[26];
  const float* h0_b1      = (const float*)d_in[27];
  const float* h0_W2      = (const float*)d_in[28];
  const float* h0_b2      = (const float*)d_in[29];
  const float* h0_W3      = (const float*)d_in[30];
  const float* h0_b3      = (const float*)d_in[31];
  float* out = (float*)d_out;
  (void)in_sizes; (void)n_in; (void)out_size; (void)ws_size;

  char* ws = (char*)d_ws;
  size_t off = 0;
  auto alloc = [&](size_t bytes) { char* p = ws + off; off = align256(off + bytes); return p; };
  float*    h       = (float*)   alloc((size_t)N_NODES * DIM * 4);
  float*    agg     = (float*)   alloc((size_t)N_NODES * DIM * 4);
  float*    t       = (float*)   alloc((size_t)N_NODES * DIM * 4);
  uint16_t* ab      = (uint16_t*)alloc((size_t)N_NODES * 320 * 2);     // bf16 A scratch
  uint16_t* wb_atom = (uint16_t*)alloc((size_t)96 * 320 * 2);
  uint16_t* wb_edge = (uint16_t*)alloc((size_t)NLAYER * 64 * 320 * 2);
  uint16_t* wb_m1   = (uint16_t*)alloc((size_t)NLAYER * 320 * 320 * 2);
  uint16_t* wb_m2   = (uint16_t*)alloc((size_t)NLAYER * 320 * 320 * 2);
  uint16_t* wb_gate = (uint16_t*)alloc((size_t)320 * 320 * 2);
  float*    gt      = (float*)   alloc((size_t)N_NODES * 4);
  float*    exv     = (float*)   alloc((size_t)N_NODES * 4);
  unsigned* gmaxu   = (unsigned*)alloc((size_t)N_GRAPH * 4);
  float*    den     = (float*)   alloc((size_t)N_GRAPH * 4);
  float*    hg      = (float*)   alloc((size_t)N_GRAPH * DIM * 4);
  float*    hcb     = (float*)   alloc((size_t)N_GRAPH * HDIM * 4);
  float*    o1      = (float*)   alloc((size_t)N_GRAPH * 2 * HDIM * 4);
  float*    o2      = (float*)   alloc((size_t)N_GRAPH * HDIM * 4);

  const long long nd = (long long)N_NODES * DIM;
  const long long na = (long long)N_NODES * 320;
  const dim3 gg(N_NODES / 16);   // 3125 blocks, 128 threads (4 waves x 5 tiles = 320 cols)

  // ---- weight / input bf16 packing (every call; deterministic) ----
  { long long tot = (long long)N_NODES * 96;
    k_pad_bf16<<<gdiv(tot, 256), 256, 0, stream>>>(x, ab, N_NODES, 92, tot, 96); }
  { long long tot = 96 * 320;
    k_pad_bf16<<<gdiv(tot, 256), 256, 0, stream>>>(atom_W, wb_atom, 92, 300, tot, 320); }
  for (int l = 0; l < NLAYER; ++l) {
    { long long tot = 64 * 320;
      k_pad_bf16<<<gdiv(tot, 256), 256, 0, stream>>>(edge_W + (size_t)l * EDIM * DIM,
          wb_edge + (size_t)l * 64 * 320, EDIM, DIM, tot, 320); }
    { long long tot = 320 * 320;
      k_pad_bf16<<<gdiv(tot, 256), 256, 0, stream>>>(mlp_W1 + (size_t)l * DIM * DIM,
          wb_m1 + (size_t)l * 320 * 320, DIM, DIM, tot, 320);
      k_pad_bf16<<<gdiv(tot, 256), 256, 0, stream>>>(mlp_W2 + (size_t)l * DIM * DIM,
          wb_m2 + (size_t)l * 320 * 320, DIM, DIM, tot, 320); }
  }
  { long long tot = 320 * 320;
    k_pad_bf16<<<gdiv(tot, 256), 256, 0, stream>>>(gate_W1, wb_gate, DIM, DIM, tot, 320); }

  // ---- atom embedding: h = x @ atom_W + atom_b ----
  k_gemm_bf16<<<gg, 128, 0, stream>>>(ab, 96, wb_atom, atom_b, h, DIM, 96, DIM, 0);

  const int* srcp = edge_index;
  const int* dstp = edge_index + N_EDGES;

  // ---- GINE layers ----
  for (int l = 0; l < NLAYER; ++l) {
    k_init_agg<<<gdiv(nd, 256), 256, 0, stream>>>(h, agg, eps, l, nd);
    k_edge_msg<<<N_EDGES / 16, 128, 0, stream>>>(edge_attr,
        wb_edge + (size_t)l * 64 * 320, edge_b + (size_t)l * DIM, h, srcp, dstp, agg);
    k_pad_bf16<<<gdiv(na, 256), 256, 0, stream>>>(agg, ab, N_NODES, DIM, na, 320);
    k_gemm_bf16<<<gg, 128, 0, stream>>>(ab, 320, wb_m1 + (size_t)l * 320 * 320,
        mlp_b1 + (size_t)l * DIM, t, DIM, 320, DIM, 1);
    k_pad_bf16<<<gdiv(na, 256), 256, 0, stream>>>(t, ab, N_NODES, DIM, na, 320);
    k_gemm_bf16<<<gg, 128, 0, stream>>>(ab, 320, wb_m2 + (size_t)l * 320 * 320,
        mlp_b2 + (size_t)l * DIM, agg, DIM, 320, DIM, 0);
    k_bn_relu<<<gdiv(nd, 256), 256, 0, stream>>>(agg, h,
        bn_gamma + (size_t)l * DIM, bn_beta + (size_t)l * DIM,
        bn_mean + (size_t)l * DIM, bn_var + (size_t)l * DIM, nd,
        (l < NLAYER - 1) ? 1 : 0);
  }

  // ---- global attention pooling ----
  k_pad_bf16<<<gdiv(na, 256), 256, 0, stream>>>(h, ab, N_NODES, DIM, na, 320);
  k_gemm_bf16<<<gg, 128, 0, stream>>>(ab, 320, wb_gate, gate_b1, t, DIM, 320, DIM, 0);
  k_gate2<<<gdiv(N_NODES, 256), 256, 0, stream>>>(t, g_bn_gamma, g_bn_beta,
      g_bn_mean, g_bn_var, gate_W2, gate_b2, gt);
  k_pool_init<<<gdiv((long long)N_GRAPH * DIM, 256), 256, 0, stream>>>(hg, gmaxu, den);
  k_gmax<<<gdiv(N_NODES, 256), 256, 0, stream>>>(gt, batch, gmaxu);
  k_ex<<<gdiv(N_NODES, 256), 256, 0, stream>>>(gt, batch, gmaxu, exv, den);
  k_pool<<<gdiv(nd, 256), 256, 0, stream>>>(h, exv, den, batch, hg, nd);

  // ---- head MLP ----
  k_hc<<<gdiv((long long)N_GRAPH * HDIM, 256), 256, 0, stream>>>(hg, gvec, hcb);
  k_dense<<<gdiv((long long)N_GRAPH * 2 * HDIM, 256), 256, 0, stream>>>(
      hcb, h0_W1, h0_b1, o1, HDIM, 2 * HDIM, 1);
  k_dense<<<gdiv((long long)N_GRAPH * HDIM, 256), 256, 0, stream>>>(
      o1, h0_W2, h0_b2, o2, 2 * HDIM, HDIM, 1);
  k_dense<<<gdiv((long long)N_GRAPH, 256), 256, 0, stream>>>(
      o2, h0_W3, h0_b3, out, HDIM, 1, 0);
}
